// GraphAttentionEncoder_69432441307176
// MI455X (gfx1250) — matrix operand order repaired and checked by reading, not compiled, not run on hardware
//
#include <hip/hip_runtime.h>
#include <stdint.h>

typedef __bf16 bf16_t;
typedef __attribute__((ext_vector_type(16))) __bf16 v16bf;
typedef __attribute__((ext_vector_type(8)))  __bf16 v8bf;
typedef __attribute__((ext_vector_type(8)))  float   v8f;

#define B_    32
#define N_    512
#define E_    1024
#define H_    8
#define L_    3
#define FF_   2048
#define KD_   128
#define DK_   16
#define TOK_  (B_ * N_)            // 16384
#define NORM_ 0.08838834764831845f // 1/sqrt(128)
#define MASK_FILL_ (-30.0f)

#define GEMM_USE_ASYNC 1           // CDNA5 global_load_async_to_lds_b128 staging

// ---------------------------------------------------------------- helpers

__device__ __forceinline__ bf16_t to_bf16(float f) {
  union { float f; unsigned u; } v; v.f = f;
  unsigned r = v.u + 0x7FFFu + ((v.u >> 16) & 1u);   // round-to-nearest-even
  union { unsigned short s; bf16_t b; } o; o.s = (unsigned short)(r >> 16);
  return o.b;
}

__device__ __forceinline__ v16bf cat16(v8bf lo, v8bf hi) {
  return __builtin_shufflevector(lo, hi, 0,1,2,3,4,5,6,7,8,9,10,11,12,13,14,15);
}

// 16B global -> LDS, ASYNCcnt-tracked (no VGPR round trip)
__device__ __forceinline__ void async_copy16(unsigned lds_addr, const bf16_t* g) {
#if GEMM_USE_ASYNC
  asm volatile("global_load_async_to_lds_b128 %0, %1, off"
               :: "v"(lds_addr), "v"(g) : "memory");
#else
  (void)lds_addr; (void)g;
#endif
}

__device__ __forceinline__ void wait_async_all() {
#if GEMM_USE_ASYNC
  asm volatile("s_wait_asynccnt 0" ::: "memory");
#endif
}

// ---------------------------------------------------------------- small kernels

__global__ void cvt_kernel(const float* __restrict__ s, bf16_t* __restrict__ d, int n) {
  int i = blockIdx.x * 256 + threadIdx.x;
  if (i < n) d[i] = to_bf16(s[i]);
}

__global__ void init_h_kernel(const float* __restrict__ x, float* __restrict__ h,
                              bf16_t* __restrict__ hb, int n) {
  int i = blockIdx.x * 256 + threadIdx.x;
  if (i < n) { float v = x[i]; h[i] = v; hb[i] = to_bf16(v); }
}

// Vb[token, h*16+d] -> Vt[((b*H+h)*16+d)*512 + n]   (keys contiguous per (b,h,d))
__global__ void transpose_v_kernel(const bf16_t* __restrict__ Vb, bf16_t* __restrict__ Vt) {
  int i = blockIdx.x * 256 + threadIdx.x;   // over TOK_*KD_
  int col = i & (KD_ - 1);
  int tok = i >> 7;
  int b = tok >> 9, n = tok & (N_ - 1);
  int h = col >> 4, d = col & 15;
  Vt[(((size_t)b * H_ + h) * DK_ + d) * N_ + n] = Vb[i];
}

__global__ void mean_kernel(const float* __restrict__ h, float* __restrict__ o) {
  int e = blockIdx.x * 256 + threadIdx.x;
  int b = blockIdx.y;
  float s = 0.f;
  const float* p = h + (size_t)b * N_ * E_ + e;
  for (int n = 0; n < N_; ++n) s += p[(size_t)n * E_];
  o[(size_t)b * E_ + e] = s * (1.0f / N_);
}

// ---------------------------------------------------------------- tiled bf16 GEMM
// C[m,n] = scale*(sum_k A[m,k]*B[n,k]) (+bias[n]) (+res[m,n]) (relu) -> outF / outB
// A: M x K row-major bf16, B: N x K row-major bf16 (weights stored [out,in]).
// Double-buffered LDS staging via CDNA5 async global->LDS copies.

#define TM 128
#define TN 128
#define TK 32
#define LDS_S 40   // 32 + 8 pad elems; row stride 80B (16B aligned)

enum { F_BIAS = 1, F_RES = 2, F_RELU = 4, F_OUTF = 8, F_OUTB = 16 };

template <int FLAGS>
__global__ __launch_bounds__(256)
void gemm_bf16_kernel(const bf16_t* __restrict__ A, const bf16_t* __restrict__ Bw,
                      const float* __restrict__ bias, const float* __restrict__ res,
                      float* __restrict__ outF, bf16_t* __restrict__ outB,
                      int M, int N, int K, float scale) {
  __shared__ alignas(16) bf16_t As[2][TM * LDS_S];
  __shared__ alignas(16) bf16_t Bs[2][TN * LDS_S];

  const int tid  = threadIdx.x;
  const int lane = tid & 31;
  const int wave = tid >> 5;
  const int wm = wave & 3;          // 4 M-strips of 32
  const int wn = wave >> 2;         // 2 N-strips of 64
  const int g  = lane >> 4;         // lane group (0/1)
  const int lm = lane & 15;

  const int blockM = blockIdx.y * TM;
  const int blockN = blockIdx.x * TN;

  const int ldRow = tid >> 1;       // 0..127
  const int ldCol = (tid & 1) * 16; // 0 / 16 (bf16 elems)

  const bf16_t* Ag = A  + (size_t)(blockM + ldRow) * K + ldCol;
  const bf16_t* Bg = Bw + (size_t)(blockN + ldRow) * K + ldCol;

  const unsigned ldsA0 = (unsigned)(uintptr_t)&As[0][ldRow * LDS_S + ldCol];
  const unsigned ldsA1 = (unsigned)(uintptr_t)&As[1][ldRow * LDS_S + ldCol];
  const unsigned ldsB0 = (unsigned)(uintptr_t)&Bs[0][ldRow * LDS_S + ldCol];
  const unsigned ldsB1 = (unsigned)(uintptr_t)&Bs[1][ldRow * LDS_S + ldCol];

#if GEMM_USE_ASYNC
  auto stage = [&](int buf, int k0) {
    unsigned la = buf ? ldsA1 : ldsA0;
    unsigned lb = buf ? ldsB1 : ldsB0;
    async_copy16(la,      Ag + k0);
    async_copy16(la + 16, Ag + k0 + 8);
    async_copy16(lb,      Bg + k0);
    async_copy16(lb + 16, Bg + k0 + 8);
  };
#else
  auto stage = [&](int buf, int k0) {
    const uint4* ap = (const uint4*)(Ag + k0);
    const uint4* bp = (const uint4*)(Bg + k0);
    uint4 a0 = ap[0], a1 = ap[1];
    uint4 b0 = bp[0], b1 = bp[1];
    *(uint4*)&As[buf][ldRow * LDS_S + ldCol]     = a0;
    *(uint4*)&As[buf][ldRow * LDS_S + ldCol + 8] = a1;
    *(uint4*)&Bs[buf][ldRow * LDS_S + ldCol]     = b0;
    *(uint4*)&Bs[buf][ldRow * LDS_S + ldCol + 8] = b1;
  };
#endif

  v8f acc[2][4] = {};

  stage(0, 0);
  wait_async_all();
  __syncthreads();

  int buf = 0;
  for (int k0 = 0; k0 < K; k0 += TK) {
    if (k0 + TK < K) stage(buf ^ 1, k0 + TK);   // prefetch next tile while computing

    // A frags: lane holds row m=lm, K = 8g+{0..7} and 8g+{16..23}
    v16bf af[2];
#pragma unroll
    for (int mt = 0; mt < 2; ++mt) {
      const bf16_t* p = &As[buf][(wm * 32 + mt * 16 + lm) * LDS_S + 8 * g];
      af[mt] = cat16(*(const v8bf*)p, *(const v8bf*)(p + 16));
    }
    // B frags: lane holds col n=lm, K = 8g+{0..7} and 8g+{16..23}
    v16bf bfv[4];
#pragma unroll
    for (int nt = 0; nt < 4; ++nt) {
      const bf16_t* p = &Bs[buf][(wn * 64 + nt * 16 + lm) * LDS_S + 8 * g];
      bfv[nt] = cat16(*(const v8bf*)p, *(const v8bf*)(p + 16));
    }
#pragma unroll
    for (int mt = 0; mt < 2; ++mt)
#pragma unroll
      for (int nt = 0; nt < 4; ++nt)
        acc[mt][nt] = __builtin_amdgcn_wmma_f32_16x16x32_bf16(
            false, af[mt], false, bfv[nt], (short)0, acc[mt][nt], false, false);

    wait_async_all();     // next tile fully landed in LDS
    __syncthreads();      // everyone done reading current tile
    buf ^= 1;
  }

  // epilogue: C layout -> VGPR r, lane L: m = r + 8*(L>>4), n = L&15
#pragma unroll
  for (int mt = 0; mt < 2; ++mt) {
#pragma unroll
    for (int nt = 0; nt < 4; ++nt) {
      const int mb = blockM + wm * 32 + mt * 16 + 8 * g;
      const int n  = blockN + wn * 64 + nt * 16 + lm;
      float bia = 0.f;
      if constexpr ((FLAGS & F_BIAS) != 0) bia = bias[n];
#pragma unroll
      for (int r = 0; r < 8; ++r) {
        const int m = mb + r;
        float v = acc[mt][nt][r] * scale + bia;
        if constexpr ((FLAGS & F_RES) != 0)  v += res[(size_t)m * N + n];
        if constexpr ((FLAGS & F_RELU) != 0) v = fmaxf(v, 0.f);
        if constexpr ((FLAGS & F_OUTF) != 0) outF[(size_t)m * N + n] = v;
        if constexpr ((FLAGS & F_OUTB) != 0) outB[(size_t)m * N + n] = to_bf16(v);
      }
    }
  }
}

// ---------------------------------------------------------------- flash attention
// grid (N/64, B, H), 128 threads (4 waves), 16 queries/wave, 32 keys/step, d=16.
__global__ __launch_bounds__(128)
void attn_kernel(const bf16_t* __restrict__ Qb, const bf16_t* __restrict__ Kb,
                 const bf16_t* __restrict__ Vt, const unsigned char* __restrict__ mask,
                 bf16_t* __restrict__ Hb) {
  __shared__ alignas(16) bf16_t P[4][16][40];   // per-wave P tile (16q x 32k), padded

  const int lane = threadIdx.x & 31;
  const int wave = threadIdx.x >> 5;
  const int g  = lane >> 4;
  const int lm = lane & 15;
  const int b  = blockIdx.y;
  const int h  = blockIdx.z;
  const int q0 = blockIdx.x * 64 + wave * 16;

  // Q fragment (Q pre-scaled by NORM): d=16 padded to K=32 with zeros
  v8bf z8 = {};
  const bf16_t* qp = Qb + ((size_t)(b * N_) + q0 + lm) * KD_ + h * DK_ + 8 * g;
  const v16bf qf = cat16(*(const v8bf*)qp, z8);

  v8f o = {};
  float rmax[8], rsum[8];
#pragma unroll
  for (int r = 0; r < 8; ++r) { rmax[r] = -1e30f; rsum[r] = 0.f; }

  const unsigned char* mp = mask + (size_t)b * N_;

  for (int kb = 0; kb < N_; kb += 32) {
    // K^T fragments: lane holds d=8g..8g+7 for key n=kb(+16)+lm in elems 0-7; elems 8-15 zero (d pad)
    v16bf kf0 = {}, kf1 = {};
    {
      const bf16_t* kp0 = Kb + ((size_t)(b * N_) + kb + lm) * KD_ + h * DK_ + 8 * g;
      kf0 = cat16(*(const v8bf*)kp0, z8);
      const bf16_t* kp1 = kp0 + (size_t)16 * KD_;
      kf1 = cat16(*(const v8bf*)kp1, z8);
    }
    v8f zc = {};
    v8f s0 = __builtin_amdgcn_wmma_f32_16x16x32_bf16(false, qf, false, kf0, (short)0, zc, false, false);
    v8f s1 = __builtin_amdgcn_wmma_f32_16x16x32_bf16(false, qf, false, kf1, (short)0, zc, false, false);

    const bool m0 = mp[kb + lm] != 0;
    const bool m1 = mp[kb + 16 + lm] != 0;

#pragma unroll
    for (int r = 0; r < 8; ++r) {
      float a = m0 ? MASK_FILL_ : s0[r];
      float c = m1 ? MASK_FILL_ : s1[r];
      float t = fmaxf(a, c);
#pragma unroll
      for (int off = 1; off < 16; off <<= 1) t = fmaxf(t, __shfl_xor(t, off, 16));
      const float nm = fmaxf(rmax[r], t);
      const float sc = __expf(rmax[r] - nm);
      const float p0 = __expf(a - nm);
      const float p1 = __expf(c - nm);
      float ps = p0 + p1;                       // masked terms stay in denominator
#pragma unroll
      for (int off = 1; off < 16; off <<= 1) ps += __shfl_xor(ps, off, 16);
      rsum[r] = rsum[r] * sc + ps;
      rmax[r] = nm;
      o[r] *= sc;
      P[wave][r + 8 * g][lm]      = to_bf16(m0 ? 0.f : p0);  // masked -> 0 in numerator
      P[wave][r + 8 * g][lm + 16] = to_bf16(m1 ? 0.f : p1);
    }
    __syncthreads();

    // P as A-fragment (16x32, full K), V as B-fragment from transposed Vt (contiguous keys)
    const bf16_t* pp = &P[wave][lm][8 * g];
    const v16bf pf = cat16(*(const v8bf*)pp, *(const v8bf*)(pp + 16));

    const bf16_t* vp = Vt + (((size_t)b * H_ + h) * DK_ + lm) * N_ + kb + 8 * g;
    const v16bf vf = cat16(*(const v8bf*)vp, *(const v8bf*)(vp + 16));

    o = __builtin_amdgcn_wmma_f32_16x16x32_bf16(false, pf, false, vf, (short)0, o, false, false);
    __syncthreads();
  }

#pragma unroll
  for (int r = 0; r < 8; ++r) {
    const float v = o[r] / rsum[r];
    Hb[((size_t)(b * N_) + q0 + r + 8 * g) * KD_ + h * DK_ + lm] = to_bf16(v);
  }
}

// ---------------------------------------------------------------- launcher

extern "C" void kernel_launch(void* const* d_in, const int* in_sizes, int n_in,
                              void* d_out, int out_size, void* d_ws, size_t ws_size,
                              hipStream_t stream) {
  (void)in_sizes; (void)n_in; (void)out_size; (void)ws_size;

  const float*         x    = (const float*)d_in[0];
  const unsigned char* mask = (const unsigned char*)d_in[1];
  const float* Wq = (const float*)d_in[2];
  const float* Wk = (const float*)d_in[3];
  const float* Wv = (const float*)d_in[4];
  const float* Wo = (const float*)d_in[5];
  const float* bo = (const float*)d_in[6];
  const float* W1 = (const float*)d_in[7];
  const float* b1 = (const float*)d_in[8];
  const float* W2 = (const float*)d_in[9];
  const float* b2 = (const float*)d_in[10];

  float* hOut    = (float*)d_out;                       // running residual stream (f32)
  float* meanOut = hOut + (size_t)TOK_ * E_;

  char* ws = (char*)d_ws;
  size_t off = 0;
  auto carve = [&](size_t bytes) -> char* {
    char* p = ws + off;
    off += (bytes + 255) & ~(size_t)255;
    return p;
  };

  const size_t nWqkv = (size_t)L_ * KD_ * E_;
  const size_t nWo   = (size_t)L_ * E_ * KD_;
  const size_t nW1   = (size_t)L_ * FF_ * E_;
  const size_t nW2   = (size_t)L_ * E_ * FF_;

  bf16_t* Wq_b = (bf16_t*)carve(nWqkv * 2);
  bf16_t* Wk_b = (bf16_t*)carve(nWqkv * 2);
  bf16_t* Wv_b = (bf16_t*)carve(nWqkv * 2);
  bf16_t* Wo_b = (bf16_t*)carve(nWo * 2);
  bf16_t* W1_b = (bf16_t*)carve(nW1 * 2);
  bf16_t* W2_b = (bf16_t*)carve(nW2 * 2);
  bf16_t* hb   = (bf16_t*)carve((size_t)TOK_ * E_ * 2);  // h in bf16
  char*   reg  = carve((size_t)TOK_ * FF_ * 2);          // 64 MiB shared region
  // FFN intermediate overlaps attention buffers (disjoint lifetimes)
  bf16_t* Fb = (bf16_t*)reg;
  bf16_t* Qb = (bf16_t*)reg;
  bf16_t* Kb = (bf16_t*)(reg + 1 * (size_t)TOK_ * KD_ * 2);
  bf16_t* Vb = (bf16_t*)(reg + 2 * (size_t)TOK_ * KD_ * 2);
  bf16_t* Vt = (bf16_t*)(reg + 3 * (size_t)TOK_ * KD_ * 2);
  bf16_t* Hb = (bf16_t*)(reg + 4 * (size_t)TOK_ * KD_ * 2);

  auto cvt = [&](const float* s, bf16_t* d, size_t n) {
    cvt_kernel<<<dim3((unsigned)((n + 255) / 256)), dim3(256), 0, stream>>>(s, d, (int)n);
  };
  cvt(Wq, Wq_b, nWqkv);
  cvt(Wk, Wk_b, nWqkv);
  cvt(Wv, Wv_b, nWqkv);
  cvt(Wo, Wo_b, nWo);
  cvt(W1, W1_b, nW1);
  cvt(W2, W2_b, nW2);

  init_h_kernel<<<dim3((TOK_ * E_) / 256), dim3(256), 0, stream>>>(x, hOut, hb, TOK_ * E_);

  auto gemm = [&](const bf16_t* A, const bf16_t* Bw, const float* bias, const float* res,
                  float* oF, bf16_t* oB, int M, int N, int K, float scale, int flags) {
    dim3 grid(N / TN, M / TM);
    if (flags == F_OUTB)
      gemm_bf16_kernel<F_OUTB><<<grid, dim3(256), 0, stream>>>(A, Bw, bias, res, oF, oB, M, N, K, scale);
    else if (flags == (F_BIAS | F_RELU | F_OUTB))
      gemm_bf16_kernel<F_BIAS | F_RELU | F_OUTB><<<grid, dim3(256), 0, stream>>>(A, Bw, bias, res, oF, oB, M, N, K, scale);
    else
      gemm_bf16_kernel<F_BIAS | F_RES | F_OUTF | F_OUTB><<<grid, dim3(256), 0, stream>>>(A, Bw, bias, res, oF, oB, M, N, K, scale);
  };

  for (int l = 0; l < L_; ++l) {
    const bf16_t* wq = Wq_b + (size_t)l * KD_ * E_;
    const bf16_t* wk = Wk_b + (size_t)l * KD_ * E_;
    const bf16_t* wv = Wv_b + (size_t)l * KD_ * E_;
    const bf16_t* wo = Wo_b + (size_t)l * E_ * KD_;
    const bf16_t* w1 = W1_b + (size_t)l * FF_ * E_;
    const bf16_t* w2 = W2_b + (size_t)l * E_ * FF_;
    const float* bol = bo + (size_t)l * E_;
    const float* b1l = b1 + (size_t)l * FF_;
    const float* b2l = b2 + (size_t)l * E_;

    // Q/K/V projections (Q pre-scaled by 1/sqrt(KD))
    gemm(hb, wq, nullptr, nullptr, nullptr, Qb, TOK_, KD_, E_, NORM_, F_OUTB);
    gemm(hb, wk, nullptr, nullptr, nullptr, Kb, TOK_, KD_, E_, 1.0f, F_OUTB);
    gemm(hb, wv, nullptr, nullptr, nullptr, Vb, TOK_, KD_, E_, 1.0f, F_OUTB);

    transpose_v_kernel<<<dim3((TOK_ * KD_) / 256), dim3(256), 0, stream>>>(Vb, Vt);

    attn_kernel<<<dim3(N_ / 64, B_, H_), dim3(128), 0, stream>>>(Qb, Kb, Vt, mask, Hb);

    // out-proj + bias + residual -> h (f32 + bf16)
    gemm(Hb, wo, bol, hOut, hOut, hb, TOK_, E_, KD_, 1.0f, F_BIAS | F_RES | F_OUTF | F_OUTB);
    // FFN1 + bias + relu -> bf16 intermediate
    gemm(hb, w1, b1l, nullptr, nullptr, Fb, TOK_, FF_, E_, 1.0f, F_BIAS | F_RELU | F_OUTB);
    // FFN2 + bias + residual -> h (f32 + bf16)
    gemm(Fb, w2, b2l, hOut, hOut, hb, TOK_, E_, FF_, 1.0f, F_BIAS | F_RES | F_OUTF | F_OUTB);
  }

  mean_kernel<<<dim3(E_ / 256, B_), dim3(256), 0, stream>>>(hOut, meanOut);
}
